// GIN_9294309229066
// MI455X (gfx1250) — matrix operand order repaired and checked by reading, not compiled, not run on hardware
//
#include <hip/hip_runtime.h>
#include <hip/hip_bf16.h>

typedef __bf16 bf16;
typedef __attribute__((ext_vector_type(16))) __bf16 v16bf;
typedef __attribute__((ext_vector_type(8)))  float  v8f;

#define LDS_PITCH 136  // 128 bf16 + 8 pad (16B) -> rows spread across banks
#define MTILE 32       // nodes per block (two 16-row WMMA M-tiles)

// ---------------- utility kernels ----------------

__global__ void zero_f32(float* p, long n) {
  long i = (long)blockIdx.x * blockDim.x + threadIdx.x;
  long stride = (long)gridDim.x * blockDim.x;
  for (; i < n; i += stride) p[i] = 0.0f;
}

__global__ void degree_kernel(const int* __restrict__ dst, float* __restrict__ deg,
                              int n_edges) {
  int e = blockIdx.x * blockDim.x + threadIdx.x;
  if (e < n_edges) unsafeAtomicAdd(&deg[dst[e]], 1.0f);
}

__global__ void invdeg_kernel(const float* __restrict__ deg, float* __restrict__ invdeg,
                              int n) {
  int i = blockIdx.x * blockDim.x + threadIdx.x;
  if (i < n) invdeg[i] = 1.0f / fmaxf(deg[i], 1.0f);
}

// one wave (32 lanes) per edge; lane handles 4 consecutive floats (float4)
__global__ void scatter_add_kernel(const float* __restrict__ x,
                                   const int* __restrict__ src,
                                   const int* __restrict__ dst,
                                   float* __restrict__ agg, int n_edges) {
  int gid = blockIdx.x * blockDim.x + threadIdx.x;
  int e = gid >> 5;
  int lane = gid & 31;
  if (e >= n_edges) return;
  int s = src[e];
  int d = dst[e];
  float4 v = ((const float4*)(x + (long)s * 128))[lane];
  float* out = agg + (long)d * 128 + lane * 4;
  unsafeAtomicAdd(out + 0, v.x);
  unsafeAtomicAdd(out + 1, v.y);
  unsafeAtomicAdd(out + 2, v.z);
  unsafeAtomicAdd(out + 3, v.w);
}

// ---------------- fused GIN MLP layer (two GEMMs) ----------------

__device__ inline void put4(v16bf& b, int i, float4 q) {
  b[i + 0] = (bf16)q.x;
  b[i + 1] = (bf16)q.y;
  b[i + 2] = (bf16)q.z;
  b[i + 3] = (bf16)q.w;
}

// A fragment for 16-bit WMMA, rows [mbase, mbase+16) of a [32][LDS_PITCH] tile.
// Layout: lanes<16 -> K={0..15}+ks*32; lanes>=16 -> +16 (contiguous, matches load_b).
__device__ inline v16bf load_a(const bf16 (*tile)[LDS_PITCH], int mbase, int laneLo,
                               int hi, int ks) {
  v16bf a;
  int kA = ks * 32 + hi * 16;
#pragma unroll
  for (int e = 0; e < 8; ++e) {
    a[e]     = tile[mbase + laneLo][kA + e];
    a[e + 8] = tile[mbase + laneLo][kA + 8 + e];
  }
  return a;
}

// B fragment: column n fixed per lane, K = ks*32 + hi*16 + e (contiguous W row)
__device__ inline v16bf load_b(const float* __restrict__ W, int n, int hi, int ks) {
  v16bf b;
  const float4* wrow = (const float4*)(W + (long)n * 128 + ks * 32 + hi * 16);
  put4(b, 0, wrow[0]);
  put4(b, 4, wrow[1]);
  put4(b, 8, wrow[2]);
  put4(b, 12, wrow[3]);
  return b;
}

// OUT = [relu]( relu( (X [+ mean_agg]) @ Wa^T + ba ) @ Wb^T + bb )
// X: n x 128 fp32, Wa: 128x128, Wb: OUT2 x 128, OUT: n x OUT2
template <int OUT2, bool HAS_AGG, bool FINAL_RELU>
__global__ __launch_bounds__(256) void gin_mlp_kernel(
    const float* __restrict__ X, const float* __restrict__ AGG,
    const float* __restrict__ INVDEG,
    const float* __restrict__ Wa, const float* __restrict__ ba,
    const float* __restrict__ Wb, const float* __restrict__ bb,
    float* __restrict__ OUT, int n_nodes) {
  __shared__ bf16 xs[MTILE][LDS_PITCH];
  __shared__ bf16 hs[MTILE][LDS_PITCH];

  const int tid = threadIdx.x;
  const int wave = tid >> 5;
  const int lane = tid & 31;
  const int laneLo = lane & 15;
  const int hi = lane >> 4;  // which half-wave (0/1)
  const int r0 = blockIdx.x * MTILE;

  // ---- Phase 0: cooperative load of 32x128 tile (+ mean aggregation), fp32->bf16
  for (int chunk = tid; chunk < MTILE * 32; chunk += 256) {
    int row = chunk >> 5;
    int c4 = chunk & 31;
    int grow = r0 + row;
    float4 v = make_float4(0.f, 0.f, 0.f, 0.f);
    if (grow < n_nodes) {
      v = ((const float4*)(X + (long)grow * 128))[c4];
      if (HAS_AGG) {
        float inv = INVDEG[grow];
        float4 a = ((const float4*)(AGG + (long)grow * 128))[c4];
        v.x += a.x * inv;
        v.y += a.y * inv;
        v.z += a.z * inv;
        v.w += a.w * inv;
      }
    }
    xs[row][c4 * 4 + 0] = (bf16)v.x;
    xs[row][c4 * 4 + 1] = (bf16)v.y;
    xs[row][c4 * 4 + 2] = (bf16)v.z;
    xs[row][c4 * 4 + 3] = (bf16)v.w;
  }
  __syncthreads();

  // ---- Phase 1: h = relu( xs @ Wa^T + ba )
  // wave w owns N-tile w; computes both M-tiles reusing the B fragment
  {
    v8f c0 = {};
    v8f c1 = {};
#pragma unroll
    for (int ks = 0; ks < 4; ++ks) {
      v16bf b = load_b(Wa, wave * 16 + laneLo, hi, ks);
      v16bf a0 = load_a(xs, 0, laneLo, hi, ks);
      v16bf a1 = load_a(xs, 16, laneLo, hi, ks);
      c0 = __builtin_amdgcn_wmma_f32_16x16x32_bf16(false, a0, false, b, (short)0, c0,
                                                   false, false);
      c1 = __builtin_amdgcn_wmma_f32_16x16x32_bf16(false, a1, false, b, (short)0, c1,
                                                   false, false);
    }
    float bias = ba[wave * 16 + laneLo];
#pragma unroll
    for (int r = 0; r < 8; ++r) {
      float v0 = fmaxf(c0[r] + bias, 0.0f);
      float v1 = fmaxf(c1[r] + bias, 0.0f);
      hs[hi * 8 + r][wave * 16 + laneLo] = (bf16)v0;
      hs[16 + hi * 8 + r][wave * 16 + laneLo] = (bf16)v1;
    }
  }
  __syncthreads();

  // ---- Phase 2: out = hs @ Wb^T + bb (OUT2 outputs), optional final relu
  if (OUT2 == 128) {
    // 8 N-tiles x 2 M-tiles -> each wave: its N-tile, both M-tiles
    v8f c0 = {};
    v8f c1 = {};
#pragma unroll
    for (int ks = 0; ks < 4; ++ks) {
      v16bf b = load_b(Wb, wave * 16 + laneLo, hi, ks);
      v16bf a0 = load_a(hs, 0, laneLo, hi, ks);
      v16bf a1 = load_a(hs, 16, laneLo, hi, ks);
      c0 = __builtin_amdgcn_wmma_f32_16x16x32_bf16(false, a0, false, b, (short)0, c0,
                                                   false, false);
      c1 = __builtin_amdgcn_wmma_f32_16x16x32_bf16(false, a1, false, b, (short)0, c1,
                                                   false, false);
    }
    float bias = bb[wave * 16 + laneLo];
#pragma unroll
    for (int r = 0; r < 8; ++r) {
      float v0 = c0[r] + bias;
      float v1 = c1[r] + bias;
      if (FINAL_RELU) {
        v0 = fmaxf(v0, 0.0f);
        v1 = fmaxf(v1, 0.0f);
      }
      int g0 = r0 + hi * 8 + r;
      int g1 = g0 + 16;
      int col = wave * 16 + laneLo;
      if (g0 < n_nodes) OUT[(long)g0 * OUT2 + col] = v0;
      if (g1 < n_nodes) OUT[(long)g1 * OUT2 + col] = v1;
    }
  } else {
    // OUT2 == 64: 4 N-tiles x 2 M-tiles = 8 tiles -> one per wave (all busy)
    const int ntile = wave & 3;
    const int mbase = (wave >> 2) * 16;
    v8f c = {};
#pragma unroll
    for (int ks = 0; ks < 4; ++ks) {
      v16bf b = load_b(Wb, ntile * 16 + laneLo, hi, ks);
      v16bf a = load_a(hs, mbase, laneLo, hi, ks);
      c = __builtin_amdgcn_wmma_f32_16x16x32_bf16(false, a, false, b, (short)0, c,
                                                  false, false);
    }
    float bias = bb[ntile * 16 + laneLo];
#pragma unroll
    for (int r = 0; r < 8; ++r) {
      float v = c[r] + bias;
      if (FINAL_RELU) v = fmaxf(v, 0.0f);
      int grow = r0 + mbase + hi * 8 + r;
      if (grow < n_nodes) OUT[(long)grow * OUT2 + ntile * 16 + laneLo] = v;
    }
  }
}

// ---------------- launch ----------------

extern "C" void kernel_launch(void* const* d_in, const int* in_sizes, int n_in,
                              void* d_out, int out_size, void* d_ws, size_t ws_size,
                              hipStream_t stream) {
  const float* features = (const float*)d_in[0];
  const int* src = (const int*)d_in[1];
  const int* dst = (const int*)d_in[2];
  const float* W1a = (const float*)d_in[3];
  const float* b1a = (const float*)d_in[4];
  const float* W1b = (const float*)d_in[5];
  const float* b1b = (const float*)d_in[6];
  const float* W2a = (const float*)d_in[7];
  const float* b2a = (const float*)d_in[8];
  const float* W2b = (const float*)d_in[9];
  const float* b2b = (const float*)d_in[10];
  const float* W3a = (const float*)d_in[11];
  const float* b3a = (const float*)d_in[12];
  const float* W3b = (const float*)d_in[13];
  const float* b3b = (const float*)d_in[14];

  const int n_nodes = in_sizes[0] / 128;
  const int n_edges = in_sizes[1];

  float* bufA = (float*)d_ws;                    // n_nodes*128 fp32
  float* agg = bufA + (size_t)n_nodes * 128;     // n_nodes*128 fp32
  float* deg = agg + (size_t)n_nodes * 128;      // n_nodes fp32
  float* invdeg = deg + n_nodes;                 // n_nodes fp32

  const int nblk = (n_nodes + MTILE - 1) / MTILE;
  const int eblk = (n_edges + 255) / 256;
  const int sblk = (int)(((long)n_edges * 32 + 255) / 256);

  // degrees (graph is fixed; compute once per call)
  zero_f32<<<256, 256, 0, stream>>>(deg, n_nodes);
  degree_kernel<<<eblk, 256, 0, stream>>>(dst, deg, n_edges);
  invdeg_kernel<<<(n_nodes + 255) / 256, 256, 0, stream>>>(deg, invdeg, n_nodes);

  // layer 1
  zero_f32<<<2048, 256, 0, stream>>>(agg, (long)n_nodes * 128);
  scatter_add_kernel<<<sblk, 256, 0, stream>>>(features, src, dst, agg, n_edges);
  gin_mlp_kernel<128, true, true><<<nblk, 256, 0, stream>>>(
      features, agg, invdeg, W1a, b1a, W1b, b1b, bufA, n_nodes);

  // layer 2 (in-place on bufA: each block touches only its own 32 rows)
  zero_f32<<<2048, 256, 0, stream>>>(agg, (long)n_nodes * 128);
  scatter_add_kernel<<<sblk, 256, 0, stream>>>(bufA, src, dst, agg, n_edges);
  gin_mlp_kernel<128, true, true><<<nblk, 256, 0, stream>>>(
      bufA, agg, invdeg, W2a, b2a, W2b, b2b, bufA, n_nodes);

  // layer 3 (plain MLP, 128 -> 128 -> 64, no agg, no final relu)
  gin_mlp_kernel<64, false, false><<<nblk, 256, 0, stream>>>(
      bufA, nullptr, nullptr, W3a, b3a, W3b, b3b, (float*)d_out, n_nodes);
}
